// LlamaDecoderLayer_26096221290609
// MI455X (gfx1250) — matrix operand order repaired and checked by reading, not compiled, not run on hardware
//
#include <hip/hip_runtime.h>

// ---------------- problem constants (from reference) ----------------
constexpr int Bc  = 4;
constexpr int Sc  = 2048;
constexpr int Hc  = 1024;
constexpr int NHc = 4;
constexpr int Dc  = Hc / NHc;      // 256
constexpr int IMc = 2048;
constexpr int Ec  = 4;
constexpr int Rc  = Bc * Sc;       // 8192 tokens
constexpr float EPSc = 1e-6f;

constexpr int LDST = 72;           // LDS row stride (64 + 8 pad) in bf16

// ---------------- WMMA types ----------------
typedef __attribute__((ext_vector_type(16))) __bf16 v16bf;
typedef __attribute__((ext_vector_type(8)))  float  v8f;

union FragBF {
    v16bf v;
    unsigned u[8];
    __bf16 e[16];
};

// =====================================================================
// GEMM:  C[M,N] f32 = A[M,K] bf16 (row-major) x Bt[N,K] bf16 (row-major)^T
// Both operands K-contiguous -> all LDS traffic is b128.
// Tile 128x128, K-step 64, 256 threads = 8 waves, wave tile 32x64.
// Latency hiding via multi-block residency per WGP (36KB LDS, ~130 VGPR)
// rather than cross-barrier register staging (which the allocator spills).
// Requires: M%128==0, N%128==0, K%64==0.
// =====================================================================
__global__ __launch_bounds__(256) void gemm_bf16_wmma(
    const __bf16* __restrict__ A, const __bf16* __restrict__ Bt,
    float* __restrict__ C, int M, int N, int K,
    long strideA, long strideB, long strideC)
{
    __shared__ __align__(16) __bf16 As[128 * LDST];
    __shared__ __align__(16) __bf16 Bs[128 * LDST];

    A  += (long)blockIdx.z * strideA;
    Bt += (long)blockIdx.z * strideB;
    C  += (long)blockIdx.z * strideC;

    const int tid  = threadIdx.x;
    const int lane = tid & 31;
    const int wave = tid >> 5;
    const int wr = (wave >> 1) * 32;   // wave row offset inside 128 tile
    const int wc = (wave & 1) * 64;    // wave col offset inside 128 tile
    const int m0 = blockIdx.y * 128;
    const int n0 = blockIdx.x * 128;

    // ---- per-thread fill offsets (32-bit, off a single base) ----
    int aoff[4], boff[4], lso[4];
    #pragma unroll
    for (int it = 0; it < 4; ++it) {
        int c   = it * 256 + tid;        // 0..1023
        int row = c >> 3;
        int col = (c & 7) << 3;
        aoff[it] = (m0 + row) * K + col;
        boff[it] = (n0 + row) * K + col;
        lso[it]  = row * LDST + col;
    }

    v8f acc[2][4];
    #pragma unroll
    for (int i = 0; i < 2; ++i)
        #pragma unroll
        for (int j = 0; j < 4; ++j)
            #pragma unroll
            for (int q = 0; q < 8; ++q) acc[i][j][q] = 0.0f;

    const int h  = lane >> 4;     // A K-half / B K-half / C row-half
    const int mm = lane & 15;     // A row within 16 / B-C column

    for (int k0 = 0; k0 < K; k0 += 64) {
        __syncthreads();
        // ---- fill: batch-issue all 8 global b128 loads, then store ----
        {
            uint4 ta[4], tb[4];
            #pragma unroll
            for (int it = 0; it < 4; ++it) {
                ta[it] = *(const uint4*)(A  + aoff[it] + k0);
                tb[it] = *(const uint4*)(Bt + boff[it] + k0);
            }
            #pragma unroll
            for (int it = 0; it < 4; ++it) {
                *(uint4*)&As[lso[it]] = ta[it];
                *(uint4*)&Bs[lso[it]] = tb[it];
            }
        }
        __syncthreads();

        // ---- compute from LDS ----
        #pragma unroll
        for (int ks = 0; ks < 64; ks += 32) {
            FragBF fa[2], fb[4];
            // A fragment (ISA 16-bit A layout): 2 x ds_load_b128 each
            #pragma unroll
            for (int mt = 0; mt < 2; ++mt) {
                const __bf16* rowp = &As[(wr + mt * 16 + mm) * LDST + ks];
                #pragma unroll
                for (int p = 0; p < 8; ++p) {
                    int kq = ((p >> 2) << 4) + (h << 3) + ((p & 3) << 1);
                    fa[mt].u[p] = *(const unsigned*)(rowp + kq);
                }
            }
            // B fragment: same K layout as A (ISA 16-bit layout) for
            // column n
            #pragma unroll
            for (int nt = 0; nt < 4; ++nt) {
                const __bf16* colp =
                    &Bs[(wc + nt * 16 + mm) * LDST + ks];
                #pragma unroll
                for (int p = 0; p < 8; ++p)
                    fb[nt].u[p] = *(const unsigned*)(colp + ((p >> 2) << 4) + (h << 3) + ((p & 3) << 1));
            }
            #pragma unroll
            for (int mt = 0; mt < 2; ++mt)
                #pragma unroll
                for (int nt = 0; nt < 4; ++nt)
                    acc[mt][nt] = __builtin_amdgcn_wmma_f32_16x16x32_bf16(
                        false, fa[mt].v, false, fb[nt].v,
                        (short)0, acc[mt][nt], false, false);
        }
    }

    // ---- store D (16x16 f32 layout: VGPR r -> M = r + 8*(lane/16)) ----
    #pragma unroll
    for (int mt = 0; mt < 2; ++mt)
        #pragma unroll
        for (int nt = 0; nt < 4; ++nt) {
            float* cp = C + (long)(m0 + wr + mt * 16 + h * 8) * N
                          + n0 + wc + nt * 16 + mm;
            #pragma unroll
            for (int r = 0; r < 8; ++r)
                cp[(long)r * N] = acc[mt][nt][r];
        }
}

// =====================================================================
// Tiled transpose + cast: src[Kd,Nd] f32 -> dst[Nd,Kd] bf16
// block (32,8), tile 32x32, both global sides coalesced.
// =====================================================================
__global__ __launch_bounds__(256) void transpose_cast_kernel(
    const float* __restrict__ src, __bf16* __restrict__ dst,
    int Kd, int Nd)
{
    __shared__ float t[32][33];
    const int n0 = blockIdx.x * 32;
    const int k0 = blockIdx.y * 32;
    const int tx = threadIdx.x, ty = threadIdx.y;
    #pragma unroll
    for (int j = 0; j < 4; ++j)
        t[ty + j * 8][tx] = src[(long)(k0 + ty + j * 8) * Nd + n0 + tx];
    __syncthreads();
    #pragma unroll
    for (int j = 0; j < 4; ++j)
        dst[(long)(n0 + ty + j * 8) * Kd + k0 + tx] =
            (__bf16)t[tx][ty + j * 8];
}

// =====================================================================
// RMSNorm + cast to bf16.  One block (256 thr) per row of Hc=1024.
// =====================================================================
__global__ __launch_bounds__(256) void rmsnorm_cast_kernel(
    const float* __restrict__ x, const float* __restrict__ w,
    __bf16* __restrict__ out)
{
    __shared__ float red[256];
    const int row = blockIdx.x;
    const int t = threadIdx.x;
    const float* xr = x + (long)row * Hc;
    float v[4];
    float ss = 0.f;
    #pragma unroll
    for (int j = 0; j < 4; ++j) {
        v[j] = xr[t + j * 256];
        ss += v[j] * v[j];
    }
    red[t] = ss; __syncthreads();
    for (int s = 128; s > 0; s >>= 1) {
        if (t < s) red[t] += red[t + s];
        __syncthreads();
    }
    float scale = rsqrtf(red[0] / Hc + EPSc);
    __bf16* orow = out + (long)row * Hc;
    #pragma unroll
    for (int j = 0; j < 4; ++j) {
        int c = t + j * 256;
        orow[c] = (__bf16)(w[c] * v[j] * scale);
    }
}

// q/k/v epilogue: [B,S,NH*D] f32 -> bf16; q,k: elu+1, [B,NH,S,D];
// v: plain, head-transposed [B,NH,D,S] (so attn@v consumes it as B^T)
__global__ void qkv_permute_kernel(const float* __restrict__ src,
                                   __bf16* __restrict__ dst,
                                   int do_elu, int trans_v)
{
    long i = (long)blockIdx.x * blockDim.x + threadIdx.x;
    if (i >= (long)Rc * Hc) return;
    int r = (int)(i / Hc), c = (int)(i % Hc);
    int b = r / Sc, s = r % Sc;
    int hh = c / Dc, d = c % Dc;
    float v = src[i];
    if (do_elu) v = (v > 0.f) ? (v + 1.f) : __expf(v);
    long head = (long)(b * NHc + hh);
    if (trans_v)
        dst[(head * Dc + d) * Sc + s] = (__bf16)v;
    else
        dst[(head * Sc + s) * Dc + d] = (__bf16)v;
}

// softmax over rows of length Sc, write bf16 probs
__global__ __launch_bounds__(256) void softmax_cast_kernel(
    const float* __restrict__ scores, __bf16* __restrict__ probs)
{
    __shared__ float red[256];
    const long row = blockIdx.x;
    const int t = threadIdx.x;
    const float* sr = scores + row * Sc;
    float v[8];
    float mx = -3.4e38f;
    #pragma unroll
    for (int j = 0; j < 8; ++j) {
        v[j] = sr[t + j * 256];
        mx = fmaxf(mx, v[j]);
    }
    red[t] = mx; __syncthreads();
    for (int s = 128; s > 0; s >>= 1) {
        if (t < s) red[t] = fmaxf(red[t], red[t + s]);
        __syncthreads();
    }
    mx = red[0]; __syncthreads();
    float sum = 0.f;
    #pragma unroll
    for (int j = 0; j < 8; ++j) { v[j] = __expf(v[j] - mx); sum += v[j]; }
    red[t] = sum; __syncthreads();
    for (int s = 128; s > 0; s >>= 1) {
        if (t < s) red[t] += red[t + s];
        __syncthreads();
    }
    float inv = 1.0f / red[0];
    __bf16* pr = probs + row * Sc;
    #pragma unroll
    for (int j = 0; j < 8; ++j) pr[t + j * 256] = (__bf16)(v[j] * inv);
}

// attention out per batch b: [NH,S,D] f32 -> ob[b*S.., NH*D] bf16
__global__ void permute_o_kernel(const float* __restrict__ of,
                                 __bf16* __restrict__ ob, int b)
{
    long i = (long)blockIdx.x * blockDim.x + threadIdx.x;
    if (i >= (long)NHc * Sc * Dc) return;
    int hh = (int)(i / ((long)Sc * Dc));
    int s  = (int)((i / Dc) % Sc);
    int d  = (int)(i % Dc);
    ob[((long)b * Sc + s) * Hc + hh * Dc + d] = (__bf16)of[i];
}

// h1 = hidden + attn_out
__global__ void add_kernel(const float* __restrict__ a,
                           const float* __restrict__ b,
                           float* __restrict__ o, long n)
{
    long i = (long)blockIdx.x * blockDim.x + threadIdx.x;
    if (i < n) o[i] = a[i] + b[i];
}

__global__ void copy_kernel(const float* __restrict__ s,
                            float* __restrict__ d, long n)
{
    long i = (long)blockIdx.x * blockDim.x + threadIdx.x;
    if (i < n) d[i] = s[i];
}

// router: logits = x2 @ W[H,E] + b ; softmax over E=4
__global__ void router_kernel(const __bf16* __restrict__ x2b,
                              const float* __restrict__ w,
                              const float* __restrict__ bias,
                              float* __restrict__ rw)
{
    int r = blockIdx.x * blockDim.x + threadIdx.x;
    if (r >= Rc) return;
    float acc[Ec];
    #pragma unroll
    for (int e = 0; e < Ec; ++e) acc[e] = bias[e];
    const __bf16* xr = x2b + (long)r * Hc;
    for (int hh = 0; hh < Hc; ++hh) {
        float xv = (float)xr[hh];
        #pragma unroll
        for (int e = 0; e < Ec; ++e) acc[e] += xv * w[hh * Ec + e];
    }
    float mx = acc[0];
    #pragma unroll
    for (int e = 1; e < Ec; ++e) mx = fmaxf(mx, acc[e]);
    float sum = 0.f;
    #pragma unroll
    for (int e = 0; e < Ec; ++e) { acc[e] = __expf(acc[e] - mx); sum += acc[e]; }
    float inv = 1.0f / sum;
    #pragma unroll
    for (int e = 0; e < Ec; ++e) rw[(long)r * Ec + e] = acc[e] * inv;
}

// balance loss: mean over batch -> mean squared deviation * 0.01
__global__ __launch_bounds__(256) void balance_loss_kernel(
    const float* __restrict__ rw, float* __restrict__ out_loss)
{
    __shared__ float red[256];
    int t = threadIdx.x;
    float acc = 0.f;
    for (int i = t; i < Sc * Ec; i += 256) {
        int s = i / Ec, e = i % Ec;
        float m = 0.f;
        #pragma unroll
        for (int b = 0; b < Bc; ++b) m += rw[((long)b * Sc + s) * Ec + e];
        m *= (1.0f / Bc);
        float d = m - 1.0f / Ec;
        acc += d * d;
    }
    red[t] = acc; __syncthreads();
    for (int s = 128; s > 0; s >>= 1) {
        if (t < s) red[t] += red[t + s];
        __syncthreads();
    }
    if (t == 0) *out_loss = red[0] / (float)(Sc * Ec) * 0.01f;
}

// silu(g) * u -> bf16
__global__ void silu_mul_kernel(const float* __restrict__ g,
                                const float* __restrict__ u,
                                __bf16* __restrict__ d, long n)
{
    long i = (long)blockIdx.x * blockDim.x + threadIdx.x;
    if (i < n) {
        float gv = g[i];
        float s = gv / (1.0f + __expf(-gv));
        d[i] = (__bf16)(s * u[i]);
    }
}

// out[r,c] += partial[r,c] * rw[r,e]
__global__ void scaled_add_kernel(float* __restrict__ out,
                                  const float* __restrict__ part,
                                  const float* __restrict__ rw, int e)
{
    long i = (long)blockIdx.x * blockDim.x + threadIdx.x;
    if (i >= (long)Rc * Hc) return;
    long r = i / Hc;
    out[i] += part[i] * rw[r * Ec + e];
}

// =====================================================================
// Host launch
// =====================================================================
static inline dim3 grid1d(long n) { return dim3((unsigned)((n + 255) / 256)); }

extern "C" void kernel_launch(void* const* d_in, const int* in_sizes, int n_in,
                              void* d_out, int out_size, void* d_ws, size_t ws_size,
                              hipStream_t stream) {
    (void)in_sizes; (void)n_in; (void)out_size; (void)ws_size;
    const float* hidden = (const float*)d_in[0];
    const float* ln1_w  = (const float*)d_in[1];
    const float* wq     = (const float*)d_in[2];
    const float* wk     = (const float*)d_in[3];
    const float* wv     = (const float*)d_in[4];
    const float* wo     = (const float*)d_in[5];
    const float* ln2_w  = (const float*)d_in[6];
    const float* rtw    = (const float*)d_in[7];
    const float* rtb    = (const float*)d_in[8];
    const float* gate_w = (const float*)d_in[9];
    const float* up_w   = (const float*)d_in[10];
    const float* down_w = (const float*)d_in[11];
    float* out = (float*)d_out;

    // ---- bump allocator over d_ws ----
    char* ws = (char*)d_ws;
    size_t off = 0;
    auto alloc = [&](size_t bytes) -> void* {
        void* p = ws + off;
        off += (bytes + 255) & ~(size_t)255;
        return p;
    };
    const long RH  = (long)Rc * Hc;
    const long RIM = (long)Rc * IMc;

    __bf16* xb     = (__bf16*)alloc(RH * 2);
    __bf16* wqbT   = (__bf16*)alloc((long)Hc * Hc * 2);   // [H,H]^T
    __bf16* wkbT   = (__bf16*)alloc((long)Hc * Hc * 2);
    __bf16* wvbT   = (__bf16*)alloc((long)Hc * Hc * 2);
    __bf16* wobT   = (__bf16*)alloc((long)Hc * Hc * 2);
    __bf16* gatebT = (__bf16*)alloc((long)Ec * Hc * IMc * 2); // per-e [IM,H]
    __bf16* upbT   = (__bf16*)alloc((long)Ec * Hc * IMc * 2); // per-e [IM,H]
    __bf16* downbT = (__bf16*)alloc((long)Ec * IMc * Hc * 2); // per-e [H,IM]
    __bf16* qb     = (__bf16*)alloc(RH * 2);   // [B,NH,S,D]
    __bf16* kb     = (__bf16*)alloc(RH * 2);   // [B,NH,S,D] (used as B^T)
    __bf16* vbT    = (__bf16*)alloc(RH * 2);   // [B,NH,D,S]
    float*  h1     = (float*)alloc(RH * 4);
    __bf16* x2b    = (__bf16*)alloc(RH * 2);
    float*  rw     = (float*)alloc((long)Rc * Ec * 4);
    __bf16* ob     = (__bf16*)alloc(RH * 2);
    float*  attn_out = (float*)alloc(RH * 4);
    // union scratch (max of QKV 96MB | ATT 109MB | MoE 201MB)
    char* uni = (char*)alloc((size_t)RIM * 4 * 2 + (size_t)RIM * 2 + (size_t)RH * 4);

    float*  qf = (float*)uni;
    float*  kf = qf + RH;
    float*  vf = kf + RH;
    float*  scores = (float*)uni;                                   // [NH,S,S]
    __bf16* attnb  = (__bf16*)(uni + (size_t)NHc * Sc * Sc * 4);    // [NH,S,S]
    float*  of     = (float*)(uni + (size_t)NHc * Sc * Sc * 4
                                  + (size_t)NHc * Sc * Sc * 2);     // [NH,S,D]
    float*  gbuf = (float*)uni;                                     // [R,IM]
    float*  ubuf = gbuf + RIM;
    __bf16* gub  = (__bf16*)(uni + (size_t)RIM * 4 * 2);
    float*  part = (float*)(uni + (size_t)RIM * 4 * 2 + (size_t)RIM * 2);

    // ---- 1. transpose-cast all GEMM B operands to bf16 [N,K] ----
    dim3 tb(32, 8);
    dim3 gHH(Hc / 32, Hc / 32);
    transpose_cast_kernel<<<gHH, tb, 0, stream>>>(wq, wqbT, Hc, Hc);
    transpose_cast_kernel<<<gHH, tb, 0, stream>>>(wk, wkbT, Hc, Hc);
    transpose_cast_kernel<<<gHH, tb, 0, stream>>>(wv, wvbT, Hc, Hc);
    transpose_cast_kernel<<<gHH, tb, 0, stream>>>(wo, wobT, Hc, Hc);
    for (int e = 0; e < Ec; ++e) {
        dim3 gHI(IMc / 32, Hc / 32);   // src [H,IM]
        dim3 gIH(Hc / 32, IMc / 32);   // src [IM,H]
        transpose_cast_kernel<<<gHI, tb, 0, stream>>>(
            gate_w + (long)e * Hc * IMc, gatebT + (long)e * IMc * Hc, Hc, IMc);
        transpose_cast_kernel<<<gHI, tb, 0, stream>>>(
            up_w + (long)e * Hc * IMc, upbT + (long)e * IMc * Hc, Hc, IMc);
        transpose_cast_kernel<<<gIH, tb, 0, stream>>>(
            down_w + (long)e * IMc * Hc, downbT + (long)e * Hc * IMc, IMc, Hc);
    }

    // ---- 2. x = rms(hidden, ln1) ----
    rmsnorm_cast_kernel<<<Rc, 256, 0, stream>>>(hidden, ln1_w, xb);

    // ---- 3. q/k/v projections (8192x1024x1024) ----
    dim3 gProj(Hc / 128, Rc / 128, 1);
    gemm_bf16_wmma<<<gProj, 256, 0, stream>>>(xb, wqbT, qf, Rc, Hc, Hc, 0, 0, 0);
    gemm_bf16_wmma<<<gProj, 256, 0, stream>>>(xb, wkbT, kf, Rc, Hc, Hc, 0, 0, 0);
    gemm_bf16_wmma<<<gProj, 256, 0, stream>>>(xb, wvbT, vf, Rc, Hc, Hc, 0, 0, 0);

    // ---- 4. feature map + head-major permute + cast ----
    qkv_permute_kernel<<<grid1d(RH), 256, 0, stream>>>(qf, qb, 1, 0);
    qkv_permute_kernel<<<grid1d(RH), 256, 0, stream>>>(kf, kb, 1, 0);
    qkv_permute_kernel<<<grid1d(RH), 256, 0, stream>>>(vf, vbT, 0, 1);

    // ---- 5. attention per batch (reuse 64MB score buffer) ----
    const long headQK = (long)Sc * Dc;
    const long headSS = (long)Sc * Sc;
    for (int b = 0; b < Bc; ++b) {
        const __bf16* qh = qb  + (long)b * NHc * headQK;
        const __bf16* kh = kb  + (long)b * NHc * headQK;
        const __bf16* vh = vbT + (long)b * NHc * headQK;
        // scores = q @ k^T : A=[S,D], Bt=[S,D]
        dim3 gS(Sc / 128, Sc / 128, NHc);
        gemm_bf16_wmma<<<gS, 256, 0, stream>>>(
            qh, kh, scores, Sc, Sc, Dc, headQK, headQK, headSS);
        softmax_cast_kernel<<<NHc * Sc, 256, 0, stream>>>(scores, attnb);
        // o = attn @ v : A=[S,S], Bt = v^T = [D,S]
        dim3 gO(Dc / 128, Sc / 128, NHc);
        gemm_bf16_wmma<<<gO, 256, 0, stream>>>(
            attnb, vh, of, Sc, Dc, Sc, headSS, headQK, headQK);
        permute_o_kernel<<<grid1d((long)NHc * Sc * Dc), 256, 0, stream>>>(of, ob, b);
    }

    // ---- 6. output projection + residual ----
    gemm_bf16_wmma<<<gProj, 256, 0, stream>>>(ob, wobT, attn_out, Rc, Hc, Hc, 0, 0, 0);
    add_kernel<<<grid1d(RH), 256, 0, stream>>>(hidden, attn_out, h1, RH);

    // ---- 7. MoE: norm, router, balance loss ----
    rmsnorm_cast_kernel<<<Rc, 256, 0, stream>>>(h1, ln2_w, x2b);
    router_kernel<<<Rc / 256, 256, 0, stream>>>(x2b, rtw, rtb, rw);
    balance_loss_kernel<<<1, 256, 0, stream>>>(rw, out + RH);

    // ---- 8. out = h1 + sum_e (silu(x2@g_e)*(x2@u_e)) @ d_e * rw_e ----
    copy_kernel<<<grid1d(RH), 256, 0, stream>>>(h1, out, RH);
    dim3 gGU(IMc / 128, Rc / 128, 1);
    dim3 gDN(Hc / 128, Rc / 128, 1);
    for (int e = 0; e < Ec; ++e) {
        const __bf16* ge = gatebT + (long)e * IMc * Hc;
        const __bf16* ue = upbT   + (long)e * IMc * Hc;
        const __bf16* de = downbT + (long)e * Hc * IMc;
        gemm_bf16_wmma<<<gGU, 256, 0, stream>>>(x2b, ge, gbuf, Rc, IMc, Hc, 0, 0, 0);
        gemm_bf16_wmma<<<gGU, 256, 0, stream>>>(x2b, ue, ubuf, Rc, IMc, Hc, 0, 0, 0);
        silu_mul_kernel<<<grid1d(RIM), 256, 0, stream>>>(gbuf, ubuf, gub, RIM);
        gemm_bf16_wmma<<<gDN, 256, 0, stream>>>(gub, de, part, Rc, Hc, IMc, 0, 0, 0);
        scaled_add_kernel<<<grid1d(RH), 256, 0, stream>>>(out, part, rw, e);
    }
}
